// MultiHeadAttention_4063039062504
// MI455X (gfx1250) — matrix operand order repaired and checked by reading, not compiled, not run on hardware
//
#include <hip/hip_runtime.h>
#include <hip/hip_bf16.h>

// Problem constants (from reference)
constexpr int Bc  = 2;
constexpr int Sc  = 2048;
constexpr int Dc  = 768;
constexpr int Hc  = 12;
constexpr int HDc = 64;        // head dim
constexpr int Mtot = Bc * Sc;  // 4096 rows for the projections

typedef __attribute__((ext_vector_type(16))) _Float16 v16h;
typedef __attribute__((ext_vector_type(8)))  _Float16 v8h;
typedef __attribute__((ext_vector_type(8)))  float    v8f;
typedef __attribute__((ext_vector_type(4)))  int      v4i;

// ---------------------------------------------------------------------------
// Async global->LDS staging (CDNA5 GLOBAL_LOAD_ASYNC_TO_LDS_B128, ASYNCcnt).
// Guarded: falls back to vector copy via VGPRs if the builtins are absent.
// Builtin signature (from hipcc diagnostic): first param is int4* in the
// global address space (AS1, printed as "__device__"); LDS side is AS3.
// ---------------------------------------------------------------------------
#if defined(__has_builtin)
#if __has_builtin(__builtin_amdgcn_global_load_async_to_lds_b128) && \
    __has_builtin(__builtin_amdgcn_s_wait_asynccnt)
#define USE_ASYNC_LDS 1
#endif
#endif
#ifndef USE_ASYNC_LDS
#define USE_ASYNC_LDS 0
#endif

#define GLOBAL_AS __attribute__((address_space(1)))
#define LDS_AS    __attribute__((address_space(3)))

// Copy 16 bytes (8 halves) global -> LDS. Flat LDS pointers carry the LDS byte
// offset in bits [31:0] (aperture rule), flat global pointers equal the 64-bit
// global address, so integer round-trip casts produce valid AS(1)/AS(3) ptrs.
__device__ __forceinline__ void stage16(_Float16* l, const _Float16* g) {
#if USE_ASYNC_LDS
    __builtin_amdgcn_global_load_async_to_lds_b128(
        (GLOBAL_AS v4i*)(unsigned long long)g,
        (LDS_AS v4i*)(unsigned)(unsigned long long)l,
        0, 0);
#else
    *(v8h*)l = *(const v8h*)g;
#endif
}

__device__ __forceinline__ void stage_wait() {
#if USE_ASYNC_LDS
    __builtin_amdgcn_s_wait_asynccnt(0);
#endif
}

// ---------------------------------------------------------------------------
// WMMA fragment loaders (wave32, 16x16x32 f16, per CDNA5 ISA layouts)
// ---------------------------------------------------------------------------

// A-matrix 16x32 f16, LDS row-major [m][k] with leading dim ld (halves).
// Lane L (0..15): row L, K = 0..7 (halves 0-7) and K = 16..23 (halves 8-15).
// Lane L+16   : row L, K = 8..15 and K = 24..31.
__device__ __forceinline__ v16h ld_frag_a(const _Float16* lds, int ld, int row0, int k0) {
    int lane = threadIdx.x & 31;
    int m    = lane & 15;
    int kb   = (lane >> 4) << 3;             // 0 or 8
    const _Float16* p = lds + (row0 + m) * ld + k0 + kb;
    v8h lo = *(const v8h*)(p);               // 16B aligned
    v8h hi = *(const v8h*)(p + 16);
    v16h a;
#pragma unroll
    for (int i = 0; i < 8; ++i) { a[i] = lo[i]; a[i + 8] = hi[i]; }
    return a;
}

// B-matrix 32x16 f16, LDS stored as lds[n][k], leading dim ld.
// Lane L (0..15): column N=L, K = 0..7 and 16..23; lane L+16: column N=L, K = 8..15 and 24..31.
__device__ __forceinline__ v16h ld_frag_b(const _Float16* lds, int ld, int n0, int k0) {
    int lane = threadIdx.x & 31;
    int n    = lane & 15;
    int kb   = (lane >> 4) << 3;             // 0 or 8
    const _Float16* p = lds + (n0 + n) * ld + k0 + kb;
    v8h lo = *(const v8h*)(p);
    v8h hi = *(const v8h*)(p + 16);
    v16h b;
#pragma unroll
    for (int i = 0; i < 8; ++i) { b[i] = lo[i]; b[i + 8] = hi[i]; }
    return b;
}

__device__ __forceinline__ v8f wmma_f16(v16h a, v16h b, v8f c) {
    return __builtin_amdgcn_wmma_f32_16x16x32_f16(false, a, false, b, (short)0, c,
                                                  false, false);
}

// Half-wave (16 lane) reductions; lanes 0-15 and 16-31 hold disjoint row sets.
__device__ __forceinline__ float hmax16(float v) {
#pragma unroll
    for (int m = 1; m < 16; m <<= 1) v = fmaxf(v, __shfl_xor(v, m, 32));
    return v;
}
__device__ __forceinline__ float hsum16(float v) {
#pragma unroll
    for (int m = 1; m < 16; m <<= 1) v += __shfl_xor(v, m, 32);
    return v;
}

// ---------------------------------------------------------------------------
// Conversions
// ---------------------------------------------------------------------------
__global__ __launch_bounds__(256) void cvt_f32_to_f16(const float* __restrict__ in,
                                                      _Float16* __restrict__ out, int n) {
    int i = blockIdx.x * 256 + threadIdx.x;
    if (i < n) out[i] = (_Float16)in[i];
}

// W [K][N] f32  ->  WT [N][K] f16  (transposed once so GEMM B-tiles stage as
// contiguous 128-bit LDS copies instead of scalar DS scatters)
__global__ __launch_bounds__(256) void cvt_w_f16_t(const float* __restrict__ in,
                                                   _Float16* __restrict__ out, int K, int N) {
    int i = blockIdx.x * 256 + threadIdx.x;
    if (i < K * N) {
        int k = i / N, n = i - k * N;
        out[(size_t)n * K + k] = (_Float16)in[i];
    }
}

// ---------------------------------------------------------------------------
// Tiled WMMA GEMM: C[M,N] = A[M,K](f16) * W[K,N](f16, given as WT[N][K]) + bias
// 128x64 block tile, 128 threads = 4 waves; wave w owns rows 32w..32w+31
// (two 16-row blocks sharing B fragments -> 16 WMMAs per 64-deep k-step).
// MODE 0: f16 head-split [B,H,S,HD]            (Q/K projections)
// MODE 2: f16 head-split transposed [B,H,HD,S] (V projection)
// MODE 1: f32 row-major [M,N] to d_out         (output projection)
// ---------------------------------------------------------------------------
template <int MODE>
__global__ __launch_bounds__(128) void gemm128(const _Float16* __restrict__ A,
                                               const _Float16* __restrict__ WT,
                                               const float* __restrict__ bias,
                                               void* __restrict__ out,
                                               int M, int N, int K) {
    __shared__ _Float16 sA[128 * 64];  // [m][k]
    __shared__ _Float16 sB[64 * 64];   // [n][k]

    const int tid  = threadIdx.x;
    const int wave = tid >> 5;
    const int lane = tid & 31;
    const int m0 = blockIdx.x * 128;
    const int n0 = blockIdx.y * 64;

    v8f c[2][4] = {{v8f{}, v8f{}, v8f{}, v8f{}}, {v8f{}, v8f{}, v8f{}, v8f{}}};

    for (int k0 = 0; k0 < K; k0 += 64) {
        __syncthreads();
        // A tile: 128x64 halves, 1024 8-half vectors / 128 threads (async -> LDS)
#pragma unroll
        for (int i = tid; i < 1024; i += 128) {
            int r = i >> 3, cc = (i & 7) << 3;
            stage16(sA + r * 64 + cc, A + (size_t)(m0 + r) * K + k0 + cc);
        }
        // B tile from pre-transposed weights: straight copy into [n][k]
#pragma unroll
        for (int i = tid; i < 512; i += 128) {
            int r = i >> 3, cc = (i & 7) << 3;
            stage16(sB + r * 64 + cc, WT + (size_t)(n0 + r) * K + k0 + cc);
        }
        stage_wait();
        __syncthreads();
#pragma unroll
        for (int kk = 0; kk < 64; kk += 32) {
            v16h a0 = ld_frag_a(sA, 64, wave * 32, kk);
            v16h a1 = ld_frag_a(sA, 64, wave * 32 + 16, kk);
#pragma unroll
            for (int nb = 0; nb < 4; ++nb) {
                v16h b = ld_frag_b(sB, 64, nb * 16, kk);
                c[0][nb] = wmma_f16(a0, b, c[0][nb]);
                c[1][nb] = wmma_f16(a1, b, c[1][nb]);
            }
        }
    }

    // Epilogue. C fragment: row m = r + 8*(lane>=16), col n = 16*nb + (lane&15).
    const int ncol = lane & 15;
#pragma unroll
    for (int half = 0; half < 2; ++half) {
        const int mrow_base = m0 + wave * 32 + half * 16 + ((lane >> 4) << 3);
#pragma unroll
        for (int nb = 0; nb < 4; ++nb) {
            const int n_g = n0 + nb * 16 + ncol;
            const float bv = bias[n_g];
#pragma unroll
            for (int r = 0; r < 8; ++r) {
                const int m_g = mrow_base + r;
                const float val = c[half][nb][r] + bv;
                if (MODE == 0) {
                    int b = m_g / Sc, s = m_g - b * Sc;
                    int h = n_g / HDc, hd = n_g - h * HDc;
                    ((_Float16*)out)[(((size_t)(b * Hc + h)) * Sc + s) * HDc + hd] =
                        (_Float16)val;
                } else if (MODE == 2) {
                    int b = m_g / Sc, s = m_g - b * Sc;
                    int h = n_g / HDc, hd = n_g - h * HDc;
                    ((_Float16*)out)[(((size_t)(b * Hc + h)) * HDc + hd) * Sc + s] =
                        (_Float16)val;
                } else {
                    ((float*)out)[(size_t)m_g * N + n_g] = val;
                }
            }
        }
    }
}

// ---------------------------------------------------------------------------
// Flash attention: grid (S/64, B*H), 128 threads = 4 waves.
// Each wave owns 16 query rows; streams 64-key KV tiles through LDS.
// Q,K f16 [B,H,S,HD]; V f16 pre-transposed [B,H,HD,S]; out f16 merged [B,S,D].
// ---------------------------------------------------------------------------
__global__ __launch_bounds__(128) void flash_attn(const _Float16* __restrict__ Q,
                                                  const _Float16* __restrict__ K,
                                                  const _Float16* __restrict__ Vt,
                                                  _Float16* __restrict__ O) {
    __shared__ _Float16 sQ[64 * 64];       // [q][d]
    __shared__ _Float16 sK[64 * 64];       // [kv][d]  (B-frag reads as [n=kv][k=d])
    __shared__ _Float16 sVt[64 * 64];      // [d][kv]  (B-frag reads as [n=d][k=kv])
    __shared__ _Float16 sP[4][16 * 64];    // per-wave P strip [m][kv]

    const int tid  = threadIdx.x;
    const int wave = tid >> 5;
    const int lane = tid & 31;
    const int q0 = blockIdx.x * 64;
    const int bh = blockIdx.y;            // b*H + h

    const size_t base = (size_t)bh * Sc * HDc;
    const _Float16* Qp  = Q  + base;
    const _Float16* Kp  = K  + base;
    const _Float16* Vtp = Vt + base;      // [HD][S] per head

    // Load Q tile once (async); fragments are re-read from LDS each iteration
    // (cheap ds_load_b128s) instead of being pinned in registers -> no spills.
#pragma unroll
    for (int i = tid; i < 512; i += 128) {
        int r = i >> 3, cc = (i & 7) << 3;
        stage16(sQ + r * 64 + cc, Qp + (size_t)(q0 + r) * HDc + cc);
    }

    float mrow[8], lrow[8];
    v8f acc[4] = {v8f{}, v8f{}, v8f{}, v8f{}};
#pragma unroll
    for (int r = 0; r < 8; ++r) { mrow[r] = -1e30f; lrow[r] = 0.0f; }
    const float scale = 0.125f;  // 1/sqrt(64)

    for (int kv0 = 0; kv0 < Sc; kv0 += 64) {
        __syncthreads();
        // Stage K tile [kv][d] and V tile [d][kv] — async 128-bit LDS loads
#pragma unroll
        for (int i = tid; i < 512; i += 128) {
            int r = i >> 3, cc = (i & 7) << 3;
            stage16(sK  + r * 64 + cc, Kp  + (size_t)(kv0 + r) * HDc + cc);
            stage16(sVt + r * 64 + cc, Vtp + (size_t)r * Sc + kv0 + cc);
        }
        stage_wait();
        __syncthreads();

        // S = Q * K^T  (f32 accumulate), then scale
        v8f s[4];
        {
            v16h aQ0 = ld_frag_a(sQ, 64, wave * 16, 0);
            v16h aQ1 = ld_frag_a(sQ, 64, wave * 16, 32);
#pragma unroll
            for (int nb = 0; nb < 4; ++nb) {
                v8f z = {};
                z = wmma_f16(aQ0, ld_frag_b(sK, 64, nb * 16, 0), z);
                z = wmma_f16(aQ1, ld_frag_b(sK, 64, nb * 16, 32), z);
#pragma unroll
                for (int r = 0; r < 8; ++r) z[r] *= scale;
                s[nb] = z;
            }
        }

        // Online softmax on the C-fragment layout
        const int mbase = (lane >> 4) << 3;
#pragma unroll
        for (int r = 0; r < 8; ++r) {
            float mloc = fmaxf(fmaxf(s[0][r], s[1][r]), fmaxf(s[2][r], s[3][r]));
            mloc = hmax16(mloc);
            float mnew = fmaxf(mrow[r], mloc);
            float alpha = __expf(mrow[r] - mnew);
            mrow[r] = mnew;
            lrow[r] *= alpha;
#pragma unroll
            for (int nb = 0; nb < 4; ++nb) acc[nb][r] *= alpha;
            float psum = 0.0f;
#pragma unroll
            for (int nb = 0; nb < 4; ++nb) {
                float p = __expf(s[nb][r] - mnew);
                s[nb][r] = p;
                psum += p;
            }
            lrow[r] += hsum16(psum);
            // Spill P (f16) to per-wave LDS strip in [m][kv] layout
#pragma unroll
            for (int nb = 0; nb < 4; ++nb)
                sP[wave][(mbase + r) * 64 + nb * 16 + (lane & 15)] = (_Float16)s[nb][r];
        }
        __syncthreads();  // make P strip visible (also fences DS before reads)

        // O += P * V
        {
            v16h aP0 = ld_frag_a(sP[wave], 64, 0, 0);
            v16h aP1 = ld_frag_a(sP[wave], 64, 0, 32);
#pragma unroll
            for (int nb = 0; nb < 4; ++nb) {
                acc[nb] = wmma_f16(aP0, ld_frag_b(sVt, 64, nb * 16, 0), acc[nb]);
                acc[nb] = wmma_f16(aP1, ld_frag_b(sVt, 64, nb * 16, 32), acc[nb]);
            }
        }
    }

    // Normalize (one reciprocal per row) and write f16 merged output [B, S, D]
    float linv[8];
#pragma unroll
    for (int r = 0; r < 8; ++r) linv[r] = 1.0f / lrow[r];

    const int h = bh % Hc, b = bh / Hc;
#pragma unroll
    for (int nb = 0; nb < 4; ++nb) {
#pragma unroll
        for (int r = 0; r < 8; ++r) {
            int m = q0 + wave * 16 + ((lane >> 4) << 3) + r;
            int dcol = nb * 16 + (lane & 15);
            float val = acc[nb][r] * linv[r];
            O[((size_t)(b * Sc + m)) * Dc + h * HDc + dcol] = (_Float16)val;
        }
    }
}

// ---------------------------------------------------------------------------
// Launch
// ---------------------------------------------------------------------------
extern "C" void kernel_launch(void* const* d_in, const int* in_sizes, int n_in,
                              void* d_out, int out_size, void* d_ws, size_t ws_size,
                              hipStream_t stream) {
    const float* x  = (const float*)d_in[0];
    const float* Wq = (const float*)d_in[1];
    const float* bq = (const float*)d_in[2];
    const float* Wk = (const float*)d_in[3];
    const float* bk = (const float*)d_in[4];
    const float* Wv = (const float*)d_in[5];
    const float* bv = (const float*)d_in[6];
    const float* Wo = (const float*)d_in[7];
    const float* bo = (const float*)d_in[8];

    char* ws = (char*)d_ws;
    size_t off = 0;
    auto carve = [&](size_t bytes) { void* p = ws + off; off += bytes; return p; };

    const size_t nX = (size_t)Mtot * Dc;             // 4096*768
    const size_t nW = (size_t)Dc * Dc;               // 768*768
    const size_t nQKV = (size_t)Bc * Hc * Sc * HDc;  // == nX

    _Float16* xh   = (_Float16*)carve(nX * 2);
    _Float16* wqt  = (_Float16*)carve(nW * 2);
    _Float16* wkt  = (_Float16*)carve(nW * 2);
    _Float16* wvt  = (_Float16*)carve(nW * 2);
    _Float16* wot  = (_Float16*)carve(nW * 2);
    _Float16* qb   = (_Float16*)carve(nQKV * 2);
    _Float16* kbuf = (_Float16*)carve(nQKV * 2);
    _Float16* vtb  = (_Float16*)carve(nQKV * 2);    // [B,H,HD,S]
    _Float16* attn = (_Float16*)carve(nX * 2);

    // fp32 -> f16 conversions (weights transposed during conversion)
    cvt_f32_to_f16<<<(int)((nX + 255) / 256), 256, 0, stream>>>(x, xh, (int)nX);
    cvt_w_f16_t<<<(int)((nW + 255) / 256), 256, 0, stream>>>(Wq, wqt, Dc, Dc);
    cvt_w_f16_t<<<(int)((nW + 255) / 256), 256, 0, stream>>>(Wk, wkt, Dc, Dc);
    cvt_w_f16_t<<<(int)((nW + 255) / 256), 256, 0, stream>>>(Wv, wvt, Dc, Dc);
    cvt_w_f16_t<<<(int)((nW + 255) / 256), 256, 0, stream>>>(Wo, wot, Dc, Dc);

    // Projections
    dim3 ggrid(Mtot / 128, Dc / 64);
    gemm128<0><<<ggrid, 128, 0, stream>>>(xh, wqt, bq, qb, Mtot, Dc, Dc);
    gemm128<0><<<ggrid, 128, 0, stream>>>(xh, wkt, bk, kbuf, Mtot, Dc, Dc);
    gemm128<2><<<ggrid, 128, 0, stream>>>(xh, wvt, bv, vtb, Mtot, Dc, Dc);

    // Flash attention -> merged f16 [B,S,D]
    flash_attn<<<dim3(Sc / 64, Bc * Hc), 128, 0, stream>>>(qb, kbuf, vtb, attn);

    // Output projection -> f32 d_out
    gemm128<1><<<ggrid, 128, 0, stream>>>(attn, wot, bo, d_out, Mtot, Dc, Dc);
}
